// Attention_23811298689422
// MI455X (gfx1250) — compile-verified, matrix operand order checked by reading, not run on hardware
//
#include <hip/hip_runtime.h>
#include <hip/hip_bf16.h>

typedef __attribute__((ext_vector_type(16))) _Float16 v16h;
typedef __attribute__((ext_vector_type(8)))  _Float16 v8h;
typedef __attribute__((ext_vector_type(8)))  float    v8f;
typedef __attribute__((ext_vector_type(4)))  float    v4f;

static constexpr int E_   = 1024;
static constexpr int H_   = 16;
static constexpr int INT_ = 1024;
static constexpr int HD_  = 64;
static constexpr int B_   = 4;
static constexpr int NQ_  = 2048;
static constexpr int NK_  = 2048;

#define DEV __device__ __forceinline__

DEV v8f wmma16(v16h a, v16h b, v8f c) {
  return __builtin_amdgcn_wmma_f32_16x16x32_f16(
      /*neg_a=*/false, a, /*neg_b=*/false, b,
      /*c_mod=*/(short)0, c, /*reuse_a=*/false, /*reuse_b=*/false);
}

// Load a 16x32 f16 A-fragment (or B-fragment from [N][K]-major storage).
// Lane l < 16: row = l, K = kb+0..7 and kb+16..23 with kb = 0.
// Lane l >=16: row = l-16, kb = 8.  (ISA 7.12.2 16-bit A-matrix layout.)
DEV v16h load_frag16(const _Float16* p, int ld, int lane) {
  const int row = lane & 15;
  const int kb  = (lane >> 4) << 3;
  const _Float16* r = p + (size_t)row * ld + kb;
  v8h lo = *(const v8h*)(r);
  v8h hi = *(const v8h*)(r + 16);
  v16h f;
#pragma unroll
  for (int j = 0; j < 8; ++j) { f[j] = lo[j]; f[j + 8] = hi[j]; }
  return f;
}

// Same fragment but source is f32 row-major; convert to f16 on the fly.
DEV v16h load_frag32(const float* p, int ld, int lane) {
  const int row = lane & 15;
  const int kb  = (lane >> 4) << 3;
  const float* r = p + (size_t)row * ld + kb;
  v4f a0 = *(const v4f*)(r);
  v4f a1 = *(const v4f*)(r + 4);
  v4f a2 = *(const v4f*)(r + 16);
  v4f a3 = *(const v4f*)(r + 20);
  v16h f;
#pragma unroll
  for (int j = 0; j < 4; ++j) {
    f[j]      = (_Float16)a0[j];
    f[j + 4]  = (_Float16)a1[j];
    f[j + 8]  = (_Float16)a2[j];
    f[j + 12] = (_Float16)a3[j];
  }
  return f;
}

// Async copy of 16 bytes/lane from global to LDS (ASYNCcnt-tracked).
DEV void async_copy_b128(unsigned lds_off, const void* gptr) {
  asm volatile("global_load_async_to_lds_b128 %0, %1, off"
               :: "v"(lds_off), "v"((unsigned long long)(uintptr_t)gptr)
               : "memory");
}
DEV void wait_async0() {
  asm volatile("s_wait_asynccnt 0x0" ::: "memory");
}

// ---------------------------------------------------------------------------
// Kernel 0: weight convert + transpose, W[K][N] f32 -> WT[N][K] f16
// ---------------------------------------------------------------------------
__global__ void wconv_kernel(const float* __restrict__ W,
                             _Float16* __restrict__ WT, int K, int N) {
  int i = blockIdx.x * blockDim.x + threadIdx.x;
  if (i >= K * N) return;
  int k = i / N;
  int n = i - k * N;
  WT[(size_t)n * K + k] = (_Float16)W[i];
}

// ---------------------------------------------------------------------------
// Kernel 1: projection GEMM. Y16 = X(f32)[8192,1024] x W + bias, stored f16.
// vmode=0: head-major [B,H,N,HD];  vmode=1: d-major transposed [B,H,HD,NK].
// One wave computes a 16x64 output tile; K loop step 32; 4 WMMAs / A-load.
// ---------------------------------------------------------------------------
__global__ __launch_bounds__(256) void proj_kernel(
    const float* __restrict__ X, const _Float16* __restrict__ WT,
    const float* __restrict__ bias, _Float16* __restrict__ Y, int vmode) {
  const int lane = threadIdx.x & 31;
  const int w    = blockIdx.x * 8 + (threadIdx.x >> 5);
  const int nt   = w & 15;   // 16 n-tiles of 64 cols
  const int mt   = w >> 4;   // 512 m-tiles of 16 rows
  const int m0   = mt * 16;
  const int n0   = nt * 64;

  v8f acc[4] = {};
  for (int k0 = 0; k0 < E_; k0 += 32) {
    v16h a = load_frag32(X + (size_t)m0 * E_ + k0, E_, lane);
#pragma unroll
    for (int c = 0; c < 4; ++c) {
      v16h b = load_frag16(WT + (size_t)(n0 + c * 16) * E_ + k0, E_, lane);
      acc[c] = wmma16(a, b, acc[c]);
    }
  }

  const int rb = (lane >> 4) << 3;
  const int ci = lane & 15;
#pragma unroll
  for (int c = 0; c < 4; ++c) {
    const int col = n0 + c * 16 + ci;
    const float bv = bias[col];
    const int hh = col >> 6, d = col & 63;
#pragma unroll
    for (int r = 0; r < 8; ++r) {
      const int gm  = m0 + rb + r;
      const int bb  = gm >> 11;      // / 2048 tokens per batch
      const int tok = gm & 2047;
      const float val = acc[c][r] + bv;
      size_t idx = vmode
          ? ((size_t)((bb * H_ + hh) * HD_ + d) * NK_ + tok)
          : ((size_t)((bb * H_ + hh) * NK_ + tok) * HD_ + d);
      Y[idx] = (_Float16)val;
    }
  }
}

// ---------------------------------------------------------------------------
// Kernel 2: flash attention. A block owns 8 consecutive q-tiles (128 rows)
// of ONE (b,h). K/V 32-kv tiles are staged into LDS with double-buffered
// global_load_async_to_lds_b128 (ASYNCcnt), shared by all 8 waves; each wave
// does S = Q*K^T (4 WMMAs), online softmax, P->LDS transpose, PV (4 WMMAs).
// ---------------------------------------------------------------------------
__global__ __launch_bounds__(256) void flash_kernel(
    const _Float16* __restrict__ QH, const _Float16* __restrict__ KH,
    const _Float16* __restrict__ VT, _Float16* __restrict__ AO) {
  __shared__ __align__(16) _Float16 kbuf[2][32 * 64];  // [kv 32][d 64]
  __shared__ __align__(16) _Float16 vbuf[2][64 * 32];  // [d 64][kv 32]
  __shared__ __align__(16) _Float16 pbuf[8][16 * 32];  // per-wave P staging

  const int tid  = threadIdx.x;
  const int lane = tid & 31;
  const int wv   = tid >> 5;
  const int bh   = blockIdx.x >> 4;        // 64 (b,h) pairs
  const int qblk = blockIdx.x & 15;        // 16 blocks of 128 q rows
  const int qt   = qblk * 8 + wv;          // this wave's 16-row q tile
  const int bb   = bh >> 4;
  const int hh   = bh & 15;

  const _Float16* Qp = QH + ((size_t)bh * NQ_ + qt * 16) * HD_;
  const _Float16* Kp = KH + (size_t)bh * NK_ * HD_;
  const _Float16* Vp = VT + (size_t)bh * HD_ * NK_;

  // Per-thread staging slices: K tile is 4KB contiguous; V tile is 64 rows
  // of 64B with stride NK_ halves.
  const int vr = tid >> 2;                 // V row (d)
  const int vc = (tid & 3) * 8;            // V col chunk (kv halves)
  const unsigned lk0 = (unsigned)(uintptr_t)&kbuf[0][tid * 8];
  const unsigned lk1 = (unsigned)(uintptr_t)&kbuf[1][tid * 8];
  const unsigned lv0 = (unsigned)(uintptr_t)&vbuf[0][vr * 32 + vc];
  const unsigned lv1 = (unsigned)(uintptr_t)&vbuf[1][vr * 32 + vc];

  // Q fragments for d=0..31 and d=32..63, with 1/sqrt(HD)=0.125 folded in.
  v16h qa0 = load_frag16(Qp, HD_, lane);
  v16h qa1 = load_frag16(Qp + 32, HD_, lane);
#pragma unroll
  for (int j = 0; j < 16; ++j) {
    qa0[j] = qa0[j] * (_Float16)0.125f;
    qa1[j] = qa1[j] * (_Float16)0.125f;
  }

  v8f o[4] = {};
  float mrow[8], lrow[8];
#pragma unroll
  for (int r = 0; r < 8; ++r) { mrow[r] = -1e30f; lrow[r] = 0.0f; }

  const int rb = (lane >> 4) << 3;
  const int ci = lane & 15;
  _Float16* pb = &pbuf[wv][0];

  // Prologue: stage tile 0 into buffer 0.
  async_copy_b128(lk0, Kp + (size_t)tid * 8);
  async_copy_b128(lv0, Vp + (size_t)vr * NK_ + vc);
  wait_async0();
  __syncthreads();

  for (int kv = 0, t = 0; kv < NK_; kv += 32, ++t) {
    const int cur = t & 1;
    // Kick off the DMA for the next tile into the other buffer (safe: the
    // barrier at the end of the previous iteration guaranteed all waves are
    // done reading it). Overlaps with the 8 WMMAs + softmax below.
    if (kv + 32 < NK_) {
      const int nkv = kv + 32;
      async_copy_b128(cur ? lk0 : lk1, Kp + (size_t)nkv * HD_ + tid * 8);
      async_copy_b128(cur ? lv0 : lv1, Vp + (size_t)vr * NK_ + nkv + vc);
    }

    // S = Q * K^T for 2 kv-subtiles x 2 d-chunks (K frags from LDS)
    const _Float16* kb = &kbuf[cur][0];
    v16h kb00 = load_frag16(kb, 64, lane);
    v16h kb01 = load_frag16(kb + 32, 64, lane);
    v16h kb10 = load_frag16(kb + 16 * 64, 64, lane);
    v16h kb11 = load_frag16(kb + 16 * 64 + 32, 64, lane);
    v8f s0 = {}, s1 = {};
    s0 = wmma16(qa0, kb00, s0);
    s0 = wmma16(qa1, kb01, s0);
    s1 = wmma16(qa0, kb10, s1);
    s1 = wmma16(qa1, kb11, s1);

    // Online softmax per query row (16 lanes per half-wave hold the 16 kv
    // columns of each subtile -> xor-reduce over 16 lanes).
#pragma unroll
    for (int r = 0; r < 8; ++r) {
      float a0 = s0[r], a1 = s1[r];
      float mx = fmaxf(a0, a1);
      mx = fmaxf(mx, __shfl_xor(mx, 1));
      mx = fmaxf(mx, __shfl_xor(mx, 2));
      mx = fmaxf(mx, __shfl_xor(mx, 4));
      mx = fmaxf(mx, __shfl_xor(mx, 8));
      float mn    = fmaxf(mrow[r], mx);
      float alpha = __expf(mrow[r] - mn);
      mrow[r] = mn;
      float p0 = __expf(a0 - mn);
      float p1 = __expf(a1 - mn);
      float rs = p0 + p1;
      rs += __shfl_xor(rs, 1);
      rs += __shfl_xor(rs, 2);
      rs += __shfl_xor(rs, 4);
      rs += __shfl_xor(rs, 8);
      lrow[r] = lrow[r] * alpha + rs;
#pragma unroll
      for (int c = 0; c < 4; ++c) o[c][r] *= alpha;
      // Stage P into LDS in row-major [16 q rows][32 kv cols] f16
      pb[(rb + r) * 32 + ci]      = (_Float16)p0;
      pb[(rb + r) * 32 + 16 + ci] = (_Float16)p1;
    }

    // Reload P as a 16x32 A-fragment and do O += P * V (V frags from LDS)
    v16h pa = load_frag16(pb, 32, lane);
#pragma unroll
    for (int c = 0; c < 4; ++c) {
      v16h vb = load_frag16(&vbuf[cur][(c * 16) * 32], 32, lane);
      o[c] = wmma16(pa, vb, o[c]);
    }

    // Our slice of the next tile must be in LDS, and the buffer we just read
    // must be safe to overwrite next iteration.
    wait_async0();
    __syncthreads();
  }

  // Normalize and write merged-head output AO[B,NQ,INT] (f16)
  float inv[8];
#pragma unroll
  for (int r = 0; r < 8; ++r) inv[r] = 1.0f / lrow[r];
#pragma unroll
  for (int c = 0; c < 4; ++c) {
#pragma unroll
    for (int r = 0; r < 8; ++r) {
      const int tok = qt * 16 + rb + r;
      const int col = hh * HD_ + c * 16 + ci;
      AO[(size_t)(bb * NQ_ + tok) * INT_ + col] = (_Float16)(o[c][r] * inv[r]);
    }
  }
}

// ---------------------------------------------------------------------------
// Kernel 3: output projection. out(f32) = AO(f16) x wo + bo.
// ---------------------------------------------------------------------------
__global__ __launch_bounds__(256) void oproj_kernel(
    const _Float16* __restrict__ A, const _Float16* __restrict__ WT,
    const float* __restrict__ bias, float* __restrict__ Y) {
  const int lane = threadIdx.x & 31;
  const int w    = blockIdx.x * 8 + (threadIdx.x >> 5);
  const int nt   = w & 15;
  const int mt   = w >> 4;
  const int m0   = mt * 16;
  const int n0   = nt * 64;

  v8f acc[4] = {};
  for (int k0 = 0; k0 < INT_; k0 += 32) {
    v16h a = load_frag16(A + (size_t)m0 * INT_ + k0, INT_, lane);
#pragma unroll
    for (int c = 0; c < 4; ++c) {
      v16h b = load_frag16(WT + (size_t)(n0 + c * 16) * INT_ + k0, INT_, lane);
      acc[c] = wmma16(a, b, acc[c]);
    }
  }

  const int rb = (lane >> 4) << 3;
  const int ci = lane & 15;
#pragma unroll
  for (int c = 0; c < 4; ++c) {
    const int col = n0 + c * 16 + ci;
    const float bv = bias[col];
#pragma unroll
    for (int r = 0; r < 8; ++r) {
      const int gm = m0 + rb + r;
      Y[(size_t)gm * E_ + col] = acc[c][r] + bv;
    }
  }
}

// ---------------------------------------------------------------------------
extern "C" void kernel_launch(void* const* d_in, const int* in_sizes, int n_in,
                              void* d_out, int out_size, void* d_ws,
                              size_t ws_size, hipStream_t stream) {
  const float* q  = (const float*)d_in[0];
  const float* k  = (const float*)d_in[1];
  const float* v  = (const float*)d_in[2];
  const float* wq = (const float*)d_in[3];
  const float* bq = (const float*)d_in[4];
  const float* wk = (const float*)d_in[5];
  const float* bk = (const float*)d_in[6];
  const float* wv = (const float*)d_in[7];
  const float* bv = (const float*)d_in[8];
  const float* wo = (const float*)d_in[9];
  const float* bo = (const float*)d_in[10];
  float* out = (float*)d_out;

  _Float16* ws = (_Float16*)d_ws;
  const size_t WSZ = (size_t)E_ * INT_;        // 1M halves per weight
  const size_t TSZ = (size_t)B_ * NQ_ * INT_;  // 8M halves per tensor
  _Float16* wqT = ws;
  _Float16* wkT = ws + 1 * WSZ;
  _Float16* wvT = ws + 2 * WSZ;
  _Float16* woT = ws + 3 * WSZ;
  _Float16* QH  = ws + 4 * WSZ;
  _Float16* KH  = QH + TSZ;
  _Float16* VT  = KH + TSZ;
  _Float16* AO  = VT + TSZ;

  dim3 blk(256);
  const int cgrid = (E_ * INT_ + 255) / 256;
  wconv_kernel<<<cgrid, blk, 0, stream>>>(wq, wqT, E_, INT_);
  wconv_kernel<<<cgrid, blk, 0, stream>>>(wk, wkT, E_, INT_);
  wconv_kernel<<<cgrid, blk, 0, stream>>>(wv, wvT, E_, INT_);
  wconv_kernel<<<cgrid, blk, 0, stream>>>(wo, woT, INT_, E_);

  // 8192 waves = 512 m-tiles x 16 n-tiles; 8 waves per block
  proj_kernel<<<1024, blk, 0, stream>>>(q, wqT, bq, QH, 0);
  proj_kernel<<<1024, blk, 0, stream>>>(k, wkT, bk, KH, 0);
  proj_kernel<<<1024, blk, 0, stream>>>(v, wvT, bv, VT, 1);

  // 1024 blocks: each owns one (b,h) x 128 q rows
  flash_kernel<<<1024, blk, 0, stream>>>(QH, KH, VT, AO);

  oproj_kernel<<<1024, blk, 0, stream>>>(AO, woT, bo, out);
}
